// ConvFunc_CGCNN_edgeMLP_13194139533632
// MI455X (gfx1250) — matrix operand order repaired and checked by reading, not compiled, not run on hardware
//
#include <hip/hip_runtime.h>
#include <hip/hip_bf16.h>

// ---------------------------------------------------------------------------
// Types for CDNA5 WMMA (wave32): 16x16x32 bf16 -> f32
// ---------------------------------------------------------------------------
typedef __attribute__((ext_vector_type(16))) __bf16 v16bf;
typedef __attribute__((ext_vector_type(8)))  float  v8f;
typedef __attribute__((ext_vector_type(4))) unsigned int u32x4;
typedef __attribute__((ext_vector_type(4))) int i32x4;
typedef __attribute__((ext_vector_type(8))) int i32x8;
typedef unsigned int   u32;
typedef unsigned short u16;

struct BfFrag { uint4 lo; uint4 hi; };

static __device__ __forceinline__ v16bf make_frag(const u16* plo, const u16* phi) {
  BfFrag f;
  f.lo = *(const uint4*)plo;   // ds_load_b128 (16B aligned)
  f.hi = *(const uint4*)phi;   // ds_load_b128 (16B aligned)
  return __builtin_bit_cast(v16bf, f);
}

static __device__ __forceinline__ u16 f2bf(float x) {
  u32 u = __builtin_bit_cast(u32, x);
  u32 r = u + 0x7FFFu + ((u >> 16) & 1u);   // round-to-nearest-even
  return (u16)(r >> 16);
}
static __device__ __forceinline__ u32 pack2(float lo, float hi) {
  return (u32)f2bf(lo) | ((u32)f2bf(hi) << 16);
}
static __device__ __forceinline__ float silu_f(float x) {
  return x / (1.0f + __expf(-x));
}
static __device__ __forceinline__ float softplus_f(float x) {
  return (x > 20.0f) ? x : log1pf(__expf(x));
}

#define F 64

// ---------------------------------------------------------------------------
// TDM: 1D async copy global -> LDS of `elems` 2-byte elements.
// D# per CDNA5 ISA ch.8: group0 = {count/flags, lds_addr, global_addr, type=2},
// group1 = {data_size=2B, tensor_dim0, tile_dim0, stride}. Tracked by TENSORcnt.
// ---------------------------------------------------------------------------
#if __has_builtin(__builtin_amdgcn_tensor_load_to_lds)
#define HAVE_TDM 1
static __device__ __forceinline__ void tdm_load_1d(u32 lds_addr, const void* gaddr,
                                                   u32 elems /* <= 65535 */) {
  unsigned long long ga = (unsigned long long)(size_t)gaddr;
  u32x4 g0;
  g0[0] = 1u;                                              // count=1 (valid user D#)
  g0[1] = lds_addr;                                        // LDS byte address
  g0[2] = (u32)ga;                                         // global_addr[31:0]
  g0[3] = (u32)((ga >> 32) & 0x01FFFFFFu) | 0x80000000u;   // addr[56:32] | type=2
  i32x8 g1;
  g1[0] = 0x00010000;                        // data_size=1 -> 2 bytes/elem
  g1[1] = (int)(elems << 16);                // tensor_dim0[15:0] (bits 63:48)
  g1[2] = (int)((elems >> 16) & 0xFFFFu) | (1 << 16);  // tensor_dim0 hi | tensor_dim1=1
  g1[3] = (int)(elems << 16);                // tile_dim0 (bits 127:112)
  g1[4] = 0;                                 // tile_dim1=0 (1D), tile_dim2=0
  g1[5] = (int)elems;                        // tensor_dim0_stride lo
  g1[6] = 0;
  g1[7] = 0;
  i32x4 z4 = {0, 0, 0, 0};
#if defined(__clang_major__) && __clang_major__ >= 23
  i32x8 z8 = {0, 0, 0, 0, 0, 0, 0, 0};
  __builtin_amdgcn_tensor_load_to_lds(g0, g1, z4, z4, z8, 0);
#else
  __builtin_amdgcn_tensor_load_to_lds(g0, g1, z4, z4, 0);
#endif
}
#else
#define HAVE_TDM 0
#endif

// ---------------------------------------------------------------------------
// zero fill
// ---------------------------------------------------------------------------
__global__ void k_zero(float* __restrict__ p, long n) {
  long i = (long)blockIdx.x * blockDim.x + threadIdx.x;
  if (i < n) p[i] = 0.0f;
}

// ---------------------------------------------------------------------------
// Pre-convert weights to bf16, transposed [n][k] for the B-fragment layout.
// ---------------------------------------------------------------------------
__global__ void k_prep_w(const float* __restrict__ We, const float* __restrict__ Wm,
                         const float* __restrict__ Ws,
                         u16* __restrict__ Webf, u16* __restrict__ Wmbf,
                         u16* __restrict__ Wsbf) {
  int i = blockIdx.x * 256 + threadIdx.x;
  if (i < 4096) {                       // W_e [64,64]
    int k = i >> 6, n = i & 63;
    Webf[n * 64 + k] = f2bf(We[i]);
  } else if (i < 4096 + 12288) {        // W_m [192,64]
    int j = i - 4096;
    int k = j >> 6, n = j & 63;
    Wmbf[n * 192 + k] = f2bf(Wm[j]);
  } else if (i < 4096 + 24576) {        // W_s [192,64]
    int j = i - 16384;
    int k = j >> 6, n = j & 63;
    Wsbf[n * 192 + k] = f2bf(Ws[j]);
  }
}

// node_feats f32 -> bf16 (paired stores)
__global__ void k_prep_nf(const float* __restrict__ NF, u16* __restrict__ NFbf,
                          long npairs) {
  long i = (long)blockIdx.x * blockDim.x + threadIdx.x;
  if (i < npairs) {
    float2 v = ((const float2*)NF)[i];
    ((u32*)NFbf)[i] = pack2(v.x, v.y);
  }
}

// ---------------------------------------------------------------------------
// BN finalize: sc = gamma*rsqrt(var+eps), sh = beta - mean*sc (64 threads)
// ---------------------------------------------------------------------------
__global__ void k_finalize(const float* __restrict__ sum, const float* __restrict__ ssq,
                           const float* __restrict__ gamma, const float* __restrict__ beta,
                           float invCnt, float* __restrict__ sc, float* __restrict__ sh) {
  int j = threadIdx.x;
  float m = sum[j] * invCnt;
  float v = ssq[j] * invCnt - m * m;
  float s = gamma[j] * rsqrtf(v + 1e-5f);
  sc[j] = s;
  sh[j] = beta[j] - m * s;
}

// ---------------------------------------------------------------------------
// Pass A: column stats of (edge_feats @ W_e + b_e); also emits bf16 copy of
// edge_feats (Xbf) as a side product. No Y materialization.
// ---------------------------------------------------------------------------
__global__ __launch_bounds__(256) void k_edge_stats(
    const float* __restrict__ X, const u16* __restrict__ Webf,
    const float* __restrict__ bias, u16* __restrict__ Xbf,
    float* __restrict__ gsum, float* __restrict__ gssq, int E) {
  __shared__ __attribute__((aligned(16))) u16 sA[128 * F];   // [row][k]
  __shared__ __attribute__((aligned(16))) u16 sW[F * F];     // [n][k]
  __shared__ float sSum[F], sSsq[F];

  int tid = threadIdx.x;
  if (tid < F) { sSum[tid] = 0.f; sSsq[tid] = 0.f; }

  for (int i = tid; i < F * 32; i += 256)        // 64x64 bf16 = 2048 uints
    ((u32*)sW)[i] = ((const u32*)Webf)[i];

  long base = (long)blockIdx.x * 128;
  bool full = (base + 128) <= (long)E;

  if (full) {
    for (int i = tid; i < 128 * 32; i += 256) {  // 128 rows x 32 uint pairs
      float2 v = ((const float2*)X)[base * 32 + i];
      u32 p = pack2(v.x, v.y);
      ((u32*)sA)[i] = p;
      ((u32*)Xbf)[base * 32 + i] = p;
    }
  } else {
    for (int i = tid; i < 128 * 32; i += 256) {
      int r = i >> 5;
      long e = base + r;
      u32 p = 0;
      if (e < E) {
        float2 v = ((const float2*)X)[base * 32 + i];
        p = pack2(v.x, v.y);
        ((u32*)Xbf)[base * 32 + i] = p;
      }
      ((u32*)sA)[i] = p;
    }
  }
  __syncthreads();

  int wave = tid >> 5, lane = tid & 31;
  int g = lane >> 4, nn = lane & 15;
  int m0 = wave * 16;

  v8f acc[4];
  { v8f z = {}; for (int i = 0; i < 4; ++i) acc[i] = z; }

#pragma unroll
  for (int ks = 0; ks < 2; ++ks) {
    int kb = ks * 32;
    const u16* ap = &sA[(m0 + nn) * F + kb + g * 8];
    v16bf af = make_frag(ap, ap + 16);
#pragma unroll
    for (int nt = 0; nt < 4; ++nt) {
      const u16* bp = &sW[(nt * 16 + nn) * F + kb + g * 8];
      v16bf bf = make_frag(bp, bp + 16);
      acc[nt] = __builtin_amdgcn_wmma_f32_16x16x32_bf16(
          false, af, false, bf, (short)0, acc[nt], false, false);
    }
  }

#pragma unroll
  for (int nt = 0; nt < 4; ++nt) {
    int n = nt * 16 + nn;
    float b = bias[n];
    float ps = 0.f, pq = 0.f;
    if (full) {
#pragma unroll
      for (int r = 0; r < 8; ++r) {
        float y = acc[nt][r] + b;
        ps += y; pq += y * y;
      }
    } else {
#pragma unroll
      for (int r = 0; r < 8; ++r) {
        long e = base + m0 + r + g * 8;
        if (e < E) { float y = acc[nt][r] + b; ps += y; pq += y * y; }
      }
    }
    atomicAdd(&sSum[n], ps);
    atomicAdd(&sSsq[n], pq);
  }
  __syncthreads();
  if (tid < F) {
    atomicAdd(&gsum[tid], sSum[tid]);
    atomicAdd(&gssq[tid], sSsq[tid]);
  }
}

// ---------------------------------------------------------------------------
// Pass B: fused per-tile pipeline:
//   (TDM issued at kernel start: Xbf tile -> sX, async)
//   chunk0: A = NFbf[src]           -> accumulate W_m/W_s chunk 0
//   chunk1: A = NFbf[dst]           -> accumulate W_m/W_s chunk 1
//   chunk2: wait TENSORcnt; recompute Y_e tile via WMMA from sX,
//           A = silu(bn(Y_e)) (bf16) -> accumulate W_m/W_s chunk 2
// Writes pre-BN Y_m / Y_s plus column stats for both.
// ---------------------------------------------------------------------------
__global__ __launch_bounds__(256) void k_msg_gemm(
    const u16* __restrict__ NFbf, const u16* __restrict__ Xbf,
    const int* __restrict__ src, const int* __restrict__ dst,
    const u16* __restrict__ Webf, const float* __restrict__ be,
    const float* __restrict__ scE, const float* __restrict__ shE,
    const u16* __restrict__ Wmbf, const float* __restrict__ bm,
    const u16* __restrict__ Wsbf, const float* __restrict__ bs,
    float* __restrict__ Ym, float* __restrict__ Ys,
    float* __restrict__ gsumM, float* __restrict__ gssqM,
    float* __restrict__ gsumS, float* __restrict__ gssqS, int E) {
  __shared__ __attribute__((aligned(16))) u16 sA[128 * F];   // A tile (bf16)
  __shared__ __attribute__((aligned(16))) u16 sX[128 * F];   // edge bf16 tile
  __shared__ __attribute__((aligned(16))) u16 sWm[F * F];    // [n][k] chunk
  __shared__ __attribute__((aligned(16))) u16 sWs[F * F];
  __shared__ int   sIdx[128];
  __shared__ float sStat[4 * F];

  int tid = threadIdx.x;
  if (tid < 4 * F) sStat[tid] = 0.f;

  long base = (long)blockIdx.x * 128;
  bool full = (base + 128) <= (long)E;

#if HAVE_TDM
  // issue-early: async DMA of this tile's bf16 edge rows into sX (16KB, 1D)
  if (tid == 0)
    tdm_load_1d((u32)(size_t)sX, Xbf + base * F, 128 * F);
#endif

  // warm L2 for the next tile's edge rows (global_prefetch_b8)
  if (tid < 128) {
    long pe = base + 128 + tid;
    if (pe < E) __builtin_prefetch(&Xbf[pe * F], 0, 1);
  }

  int wave = tid >> 5, lane = tid & 31;
  int g = lane >> 4, nn = lane & 15;
  int m0 = wave * 16;

  v8f accM[4], accS[4];
  { v8f z = {}; for (int i = 0; i < 4; ++i) { accM[i] = z; accS[i] = z; } }

  for (int chunk = 0; chunk < 3; ++chunk) {
    __syncthreads();   // protect LDS reuse (and sStat init on first pass)

    if (chunk < 2) {
      if (tid < 128) {
        long e = base + tid;
        int idx = 0;
        if (e < E) idx = (chunk == 0) ? src[e] : dst[e];
        sIdx[tid] = idx;
      }
      // stage this K-chunk of both weight matrices (bf16, direct uint copy)
      for (int i = tid; i < F * 32; i += 256) {
        int n = i >> 5, kk = i & 31;
        ((u32*)sWm)[n * 32 + kk] = ((const u32*)Wmbf)[n * 96 + chunk * 32 + kk];
        ((u32*)sWs)[n * 32 + kk] = ((const u32*)Wsbf)[n * 96 + chunk * 32 + kk];
      }
      __syncthreads();
      // gather node rows (bf16, uint pairs) -- L2 resident
      for (int i = tid; i < 128 * 32; i += 256) {
        int r = i >> 5, c = i & 31;
        ((u32*)sA)[i] = ((const u32*)NFbf)[(long)sIdx[r] * 32 + c];
      }
      __syncthreads();
    } else {
      // ---- recompute Y_e tile and build A = silu(bn(Y_e)) ----
#if !HAVE_TDM
      for (int i = tid; i < 128 * 32; i += 256)
        ((u32*)sX)[i] = ((const u32*)Xbf)[base * 32 + i];
#endif
      for (int i = tid; i < F * 32; i += 256)
        ((u32*)sWm)[i] = ((const u32*)Webf)[i];          // borrow sWm for W_e
#if HAVE_TDM
      if (tid == 0) __builtin_amdgcn_s_wait_tensorcnt(0);  // wait-late
#endif
      __syncthreads();

      v8f accE[4];
      { v8f z = {}; for (int i = 0; i < 4; ++i) accE[i] = z; }
#pragma unroll
      for (int ks = 0; ks < 2; ++ks) {
        int kb = ks * 32;
        const u16* ap = &sX[(m0 + nn) * F + kb + g * 8];
        v16bf af = make_frag(ap, ap + 16);
#pragma unroll
        for (int nt = 0; nt < 4; ++nt) {
          const u16* bp = &sWm[(nt * 16 + nn) * F + kb + g * 8];
          v16bf bf = make_frag(bp, bp + 16);
          accE[nt] = __builtin_amdgcn_wmma_f32_16x16x32_bf16(
              false, af, false, bf, (short)0, accE[nt], false, false);
        }
      }
#pragma unroll
      for (int nt = 0; nt < 4; ++nt) {
        int n = nt * 16 + nn;
        float b = be[n], sc = scE[n], sh = shE[n];
#pragma unroll
        for (int r = 0; r < 8; ++r) {
          int m = m0 + r + g * 8;
          float y = (accE[nt][r] + b) * sc + sh;
          sA[m * F + n] = f2bf(silu_f(y));
        }
      }
      __syncthreads();
      // stage chunk-2 weights (overwrite W_e)
      for (int i = tid; i < F * 32; i += 256) {
        int n = i >> 5, kk = i & 31;
        ((u32*)sWm)[n * 32 + kk] = ((const u32*)Wmbf)[n * 96 + 64 + kk];
        ((u32*)sWs)[n * 32 + kk] = ((const u32*)Wsbf)[n * 96 + 64 + kk];
      }
      __syncthreads();
    }

    // ---- dual GEMM accumulate over this chunk ----
#pragma unroll
    for (int ks = 0; ks < 2; ++ks) {
      int kb = ks * 32;
      const u16* ap = &sA[(m0 + nn) * F + kb + g * 8];
      v16bf af = make_frag(ap, ap + 16);
#pragma unroll
      for (int nt = 0; nt < 4; ++nt) {
        const u16* bpm = &sWm[(nt * 16 + nn) * F + kb + g * 8];
        v16bf bfm = make_frag(bpm, bpm + 16);
        accM[nt] = __builtin_amdgcn_wmma_f32_16x16x32_bf16(
            false, af, false, bfm, (short)0, accM[nt], false, false);
        const u16* bps = &sWs[(nt * 16 + nn) * F + kb + g * 8];
        v16bf bfs = make_frag(bps, bps + 16);
        accS[nt] = __builtin_amdgcn_wmma_f32_16x16x32_bf16(
            false, af, false, bfs, (short)0, accS[nt], false, false);
      }
    }
  }

  // epilogue: bias, store pre-BN Y_m/Y_s, column stats (fast path for full tiles)
#pragma unroll
  for (int nt = 0; nt < 4; ++nt) {
    int n = nt * 16 + nn;
    float bM = bm[n], bS = bs[n];
    float psM = 0.f, pqM = 0.f, psS = 0.f, pqS = 0.f;
    if (full) {
#pragma unroll
      for (int r = 0; r < 8; ++r) {
        long e = base + m0 + r + g * 8;
        float ym = accM[nt][r] + bM;
        float ys = accS[nt][r] + bS;
        Ym[e * F + n] = ym;
        Ys[e * F + n] = ys;
        psM += ym; pqM += ym * ym;
        psS += ys; pqS += ys * ys;
      }
    } else {
#pragma unroll
      for (int r = 0; r < 8; ++r) {
        long e = base + m0 + r + g * 8;
        if (e < E) {
          float ym = accM[nt][r] + bM;
          float ys = accS[nt][r] + bS;
          Ym[e * F + n] = ym;
          Ys[e * F + n] = ys;
          psM += ym; pqM += ym * ym;
          psS += ys; pqS += ys * ys;
        }
      }
    }
    atomicAdd(&sStat[n],         psM);
    atomicAdd(&sStat[F + n],     pqM);
    atomicAdd(&sStat[2 * F + n], psS);
    atomicAdd(&sStat[3 * F + n], pqS);
  }
  __syncthreads();
  if (tid < F) {
    atomicAdd(&gsumM[tid], sStat[tid]);
    atomicAdd(&gssqM[tid], sStat[F + tid]);
    atomicAdd(&gsumS[tid], sStat[2 * F + tid]);
    atomicAdd(&gssqS[tid], sStat[3 * F + tid]);
  }
}

// ---------------------------------------------------------------------------
// Pass C: msg = silu(bn(Y_m)) * softplus(bn(Y_s)); scatter-add into agg[dst].
// ---------------------------------------------------------------------------
__global__ void k_msg_apply(const float* __restrict__ Ym, const float* __restrict__ Ys,
                            const float* __restrict__ scM, const float* __restrict__ shM,
                            const float* __restrict__ scS, const float* __restrict__ shS,
                            const int* __restrict__ dst, float* __restrict__ agg,
                            long total) {
  long i = (long)blockIdx.x * blockDim.x + threadIdx.x;
  if (i >= total) return;
  int  j = (int)(i & 63);
  long e = i >> 6;
  float m = silu_f(Ym[i] * scM[j] + shM[j]);
  float s = softplus_f(Ys[i] * scS[j] + shS[j]);
  atomicAdd(&agg[(long)dst[e] * F + j], m * s);   // global_atomic_add_f32
}

// ---------------------------------------------------------------------------
// Pass D: column stats over agg (N rows).
// ---------------------------------------------------------------------------
__global__ __launch_bounds__(256) void k_agg_stats(const float* __restrict__ agg,
                                                   float* __restrict__ gsum,
                                                   float* __restrict__ gssq, long total) {
  __shared__ float sSum[F], sSsq[F];
  int tid = threadIdx.x;
  if (tid < F) { sSum[tid] = 0.f; sSsq[tid] = 0.f; }
  __syncthreads();
  float ps = 0.f, pq = 0.f;
  for (long i = (long)blockIdx.x * 256 + tid; i < total; i += (long)gridDim.x * 256) {
    float v = agg[i];
    ps += v; pq += v * v;
  }
  atomicAdd(&sSum[tid & 63], ps);
  atomicAdd(&sSsq[tid & 63], pq);
  __syncthreads();
  if (tid < F) {
    atomicAdd(&gsum[tid], sSum[tid]);
    atomicAdd(&gssq[tid], sSsq[tid]);
  }
}

// ---------------------------------------------------------------------------
// Pass E: out = softplus(bn(agg) + node_feats)
// ---------------------------------------------------------------------------
__global__ void k_final(const float* __restrict__ agg, const float* __restrict__ NF,
                        const float* __restrict__ scN, const float* __restrict__ shN,
                        float* __restrict__ out, long total) {
  long i = (long)blockIdx.x * blockDim.x + threadIdx.x;
  if (i >= total) return;
  int j = (int)(i & 63);
  float x = agg[i] * scN[j] + shN[j] + NF[i];
  out[i] = softplus_f(x);
}

// ---------------------------------------------------------------------------
// Host-side launch
// ---------------------------------------------------------------------------
extern "C" void kernel_launch(void* const* d_in, const int* in_sizes, int n_in,
                              void* d_out, int out_size, void* d_ws, size_t ws_size,
                              hipStream_t stream) {
  const float* node_feats = (const float*)d_in[0];
  const float* edge_feats = (const float*)d_in[1];
  const int*   src        = (const int*)d_in[2];
  const int*   dst        = (const int*)d_in[3];
  const float* W_e  = (const float*)d_in[4];
  const float* b_e  = (const float*)d_in[5];
  const float* g_e  = (const float*)d_in[6];
  const float* be_e = (const float*)d_in[7];
  const float* W_m  = (const float*)d_in[8];
  const float* b_m  = (const float*)d_in[9];
  const float* g_m  = (const float*)d_in[10];
  const float* be_m = (const float*)d_in[11];
  const float* W_s  = (const float*)d_in[12];
  const float* b_s  = (const float*)d_in[13];
  const float* g_s  = (const float*)d_in[14];
  const float* be_s = (const float*)d_in[15];
  const float* g_n  = (const float*)d_in[16];
  const float* be_n = (const float*)d_in[17];

  const int N = in_sizes[0] / F;
  const int E = in_sizes[2];

  // ---- workspace layout ----
  float* ws   = (float*)d_ws;
  float* sumE = ws;        float* ssqE = ws + 64;
  float* sumM = ws + 128;  float* ssqM = ws + 192;
  float* sumS = ws + 256;  float* ssqS = ws + 320;
  float* sumN = ws + 384;  float* ssqN = ws + 448;
  float* scE  = ws + 512;  float* shE  = ws + 576;
  float* scM  = ws + 640;  float* shM  = ws + 704;
  float* scS  = ws + 768;  float* shS  = ws + 832;
  float* scN  = ws + 896;  float* shN  = ws + 960;

  u16* b16   = (u16*)(ws + 1024);
  u16* Webf  = b16;                       // 4096
  u16* Wmbf  = b16 + 4096;                // 12288
  u16* Wsbf  = b16 + 16384;               // 12288
  u16* NFbf  = b16 + 28672;               // N*64
  u16* Xbf   = NFbf + (size_t)N * F;      // E*64

  float* Ym  = (float*)(Xbf + (size_t)E * F);
  float* Ys  = Ym + (size_t)E * F;
  float* agg = Ys + (size_t)E * F;

  const long totE = (long)E * F;
  const long totN = (long)N * F;

  // zero stats + agg
  k_zero<<<4, 256, 0, stream>>>(ws, 1024);
  k_zero<<<(unsigned)((totN + 255) / 256), 256, 0, stream>>>(agg, totN);

  // precision prep: bf16 weights (transposed) + bf16 node feats
  k_prep_w<<<112, 256, 0, stream>>>(W_e, W_m, W_s, Webf, Wmbf, Wsbf);
  k_prep_nf<<<(unsigned)((totN / 2 + 255) / 256), 256, 0, stream>>>(
      node_feats, NFbf, totN / 2);

  const unsigned gemmBlocks = (unsigned)((E + 127) / 128);

  // Pass A: edge-BN stats (+ bf16 edge copy)
  k_edge_stats<<<gemmBlocks, 256, 0, stream>>>(edge_feats, Webf, b_e, Xbf,
                                               sumE, ssqE, E);
  k_finalize<<<1, 64, 0, stream>>>(sumE, ssqE, g_e, be_e, 1.0f / (float)E, scE, shE);

  // Pass B: fused recompute + gather + dual GEMM (TDM-staged edge tile)
  k_msg_gemm<<<gemmBlocks, 256, 0, stream>>>(NFbf, Xbf, src, dst,
                                             Webf, b_e, scE, shE,
                                             Wmbf, b_m, Wsbf, b_s,
                                             Ym, Ys, sumM, ssqM, sumS, ssqS, E);
  k_finalize<<<1, 64, 0, stream>>>(sumM, ssqM, g_m, be_m, 1.0f / (float)E, scM, shM);
  k_finalize<<<1, 64, 0, stream>>>(sumS, ssqS, g_s, be_s, 1.0f / (float)E, scS, shS);

  // Pass C: gated message + scatter
  k_msg_apply<<<(unsigned)((totE + 255) / 256), 256, 0, stream>>>(
      Ym, Ys, scM, shM, scS, shS, dst, agg, totE);

  // Pass D/E: node BN + output
  k_agg_stats<<<512, 256, 0, stream>>>(agg, sumN, ssqN, totN);
  k_finalize<<<1, 64, 0, stream>>>(sumN, ssqN, g_n, be_n, 1.0f / (float)N, scN, shN);
  k_final<<<(unsigned)((totN + 255) / 256), 256, 0, stream>>>(
      agg, node_feats, scN, shN, (float*)d_out, totN);
}
